// HGAHead_88837103550735
// MI455X (gfx1250) — matrix operand order repaired and checked by reading, not compiled, not run on hardware
//
#include <hip/hip_runtime.h>

// ---------------------------------------------------------------------------
// Problem constants (match reference)
// ---------------------------------------------------------------------------
#define BATCH   4
#define SEQ     1024
#define HID     768
#define NLAB    8
#define IDIM    64               // inner dim per label (q or k chunk)
#define NEGBIG  1000000000000.0f

typedef __attribute__((ext_vector_type(16))) _Float16 v16h;
typedef __attribute__((ext_vector_type(8)))  _Float16 v8h;
typedef __attribute__((ext_vector_type(2)))  _Float16 v2h;
typedef __attribute__((ext_vector_type(8)))  float    v8f;

// ---------------------------------------------------------------------------
// K0: one-shot f32 -> f16 conversion (x and W), 8 elements / thread
//   (each source element converted exactly once; K2 then reuses f16 copies)
// ---------------------------------------------------------------------------
__global__ __launch_bounds__(256) void cvt_f16_kernel(const float* __restrict__ src,
                                                      _Float16* __restrict__ dst,
                                                      int n8) {
    const int i = blockIdx.x * blockDim.x + threadIdx.x;
    if (i >= n8) return;
    const float* s = src + (size_t)i * 8;
    v8h d;
#pragma unroll
    for (int j = 0; j < 8; ++j) d[j] = (_Float16)s[j];
    *(v8h*)(dst + (size_t)i * 8) = d;          // 16B store, b128
}

// ---------------------------------------------------------------------------
// K1: box run-ids (block-wide scan) + RoPE sin/cos tables
//   cosT/sinT: [BATCH][SEQ][32] f32
// ---------------------------------------------------------------------------
__global__ __launch_bounds__(1024) void box_rope_kernel(const int* __restrict__ boxes,
                                                        float* __restrict__ cosT,
                                                        float* __restrict__ sinT) {
    __shared__ int sc[SEQ];
    const int b = blockIdx.x;
    const int t = threadIdx.x;

    const int* bp = boxes + ((size_t)b * SEQ + t) * 4;
    int change = 0;
    if (t > 0) {
        const int* pp = bp - 4;
        change = (bp[0] != pp[0]) | (bp[1] != pp[1]) | (bp[2] != pp[2]) | (bp[3] != pp[3]);
    }
    sc[t] = change;
    __syncthreads();
    // Hillis-Steele inclusive scan -> run_id[t] = #changes in (0, t]
    for (int off = 1; off < SEQ; off <<= 1) {
        int v = (t >= off) ? sc[t - off] : 0;
        __syncthreads();
        sc[t] += v;
        __syncthreads();
    }
    const int run  = sc[t];
    const int last = sc[SEQ - 1];
    const float id = (run == last) ? 0.0f : (float)run;

    // inv_freq_i = 10000^(-2i/64) = exp(-i * ln(1e4)/32)
    const float c = __logf(10000.0f) * (1.0f / 32.0f);
    float* cb = cosT + ((size_t)b * SEQ + t) * 32;
    float* sb = sinT + ((size_t)b * SEQ + t) * 32;
#pragma unroll
    for (int i = 0; i < 32; ++i) {
        float ang = id * __expf(-c * (float)i);
        sb[i] = __sinf(ang);
        cb[i] = __cosf(ang);
    }
}

// ---------------------------------------------------------------------------
// CDNA5 16-bit WMMA fragment layouts (wave32):
//   A (16x32, MxK): lane<16 -> M=lane,  halves = K{ka..ka+15}, ka = 16*(lane>>4) (same k order as B)
//   B (32x16, KxN): lane<16 -> N=lane,  halves = K{kb..kb+15},          kb = 16*(lane>>4); used here, not ISA's
//   C (16x16 f32):  lane<16 -> N=lane, rows M=v (v=0..7); lane>=16 -> M=v+8
// ---------------------------------------------------------------------------

// ---------------------------------------------------------------------------
// K2: out = x @ W^T + bias, then RoPE-rotate, pack f16 into Q/K buffers
//   Q,K layout: [BATCH][NLAB][SEQ][IDIM] f16
//   wave tile = 16 rows (m) x 64 cols (= exactly one q or k chunk of one label)
// ---------------------------------------------------------------------------
__global__ __launch_bounds__(256) void proj_rope_kernel(const _Float16* __restrict__ Xh,
                                                        const _Float16* __restrict__ Wh,
                                                        const float* __restrict__ bias,
                                                        const float* __restrict__ cosT,
                                                        const float* __restrict__ sinT,
                                                        _Float16* __restrict__ Qh,
                                                        _Float16* __restrict__ Kh) {
    __shared__ float tile[8][16 * 64];   // 8 waves * 4KB = 32KB
    const int lane = threadIdx.x & 31;
    const int wave = threadIdx.x >> 5;
    const int wt   = blockIdx.x * 8 + wave;      // 0..4095
    const int mtile = wt >> 4;                   // 0..255  (16-row tiles over B*S)
    const int chunk = wt & 15;                   // 0..15   (64-col output chunk)
    const int m0 = mtile * 16;
    const int o0 = chunk * 64;
    const int b  = m0 >> 10;
    const int s0 = m0 & (SEQ - 1);

    const int l15  = lane & 15;
    const int aoff = (lane >> 4) << 3;           // 0 or 8
    const int boff = (lane >> 4) << 4;           // 0 or 16

    v8f acc[4] = {v8f{}, v8f{}, v8f{}, v8f{}};

    const _Float16* xrow = Xh + ((size_t)(m0 + l15)) * HID;

    for (int k0 = 0; k0 < HID; k0 += 32) {
        // ---- A fragment (two 16B contiguous runs -> global_load_b128) ----
        v16h a;
        const int b1 = k0 + boff;
#pragma unroll
        for (int j = 0; j < 8; ++j) {
            a[j]     = xrow[b1 + j];
            a[j + 8] = xrow[b1 + 8 + j];
        }
#pragma unroll
        for (int nt = 0; nt < 4; ++nt) {
            // ---- B fragment: B[k][n] = W[n][k]; lane holds 16 contiguous k ----
            const int n = o0 + nt * 16 + l15;
            const _Float16* wp = Wh + (size_t)n * HID + k0 + boff;
            __builtin_prefetch(wp + 64, 0, 1);   // global_prefetch_b8 (speculative)
            v16h bf;
#pragma unroll
            for (int j = 0; j < 16; ++j) bf[j] = wp[j];
            acc[nt] = __builtin_amdgcn_wmma_f32_16x16x32_f16(
                /*neg_a=*/false, a, /*neg_b=*/false, bf,
                /*c_mod=*/(short)0, acc[nt], /*reuse_a=*/false, /*reuse_b=*/false);
        }
    }

    // ---- stage the 16x64 f32 tile through LDS (C layout is column-per-lane) ----
    float* t = tile[wave];
#pragma unroll
    for (int nt = 0; nt < 4; ++nt)
#pragma unroll
        for (int v = 0; v < 8; ++v)
            t[(v + aoff) * 64 + nt * 16 + l15] = acc[nt][v];
    __syncthreads();

    // ---- RoPE rotation: lane handles column pair (2*lane, 2*lane+1) ----
    const int c0 = lane * 2;
    const int hlab = chunk >> 1;
    const int isK  = chunk & 1;
    _Float16* outBase = (isK ? Kh : Qh) + ((size_t)(b * NLAB + hlab)) * SEQ * IDIM;
    const float bi0 = bias[o0 + c0];
    const float bi1 = bias[o0 + c0 + 1];

#pragma unroll 4
    for (int r = 0; r < 16; ++r) {
        const int s = s0 + r;
        float t0 = t[r * 64 + c0]     + bi0;
        float t1 = t[r * 64 + c0 + 1] + bi1;
        const size_t cs = ((size_t)b * SEQ + s) * 32 + lane;
        const float cv = cosT[cs];
        const float sv = sinT[cs];
        const float r0 = t0 * cv - t1 * sv;
        const float r1 = t1 * cv + t0 * sv;
        v2h pk;
        pk[0] = (_Float16)r0;
        pk[1] = (_Float16)r1;
        *(v2h*)(outBase + (size_t)s * IDIM + c0) = pk;
    }
}

// ---------------------------------------------------------------------------
// K3: logits[b,h,m,n] = (q . k)/8 with pad + causal masking
//   wave tile = 16 (m) x 64 (n), K = 64 in two 32-chunks -> 8 WMMAs / wave
// ---------------------------------------------------------------------------
__global__ __launch_bounds__(256) void logits_kernel(const _Float16* __restrict__ Qh,
                                                     const _Float16* __restrict__ Kh,
                                                     const float* __restrict__ amask,
                                                     float* __restrict__ out) {
    const int lane = threadIdx.x & 31;
    const int wave = threadIdx.x >> 5;
    const int wt   = blockIdx.x * 8 + wave;      // 0..32767
    const int nt64 = wt & 15;                    // 16 n-tiles of 64
    const int mt   = (wt >> 4) & 63;             // 64 m-tiles of 16
    const int bh   = wt >> 10;                   // 0..31  (= b*8 + h)
    const int b    = bh >> 3;
    const int m0 = mt * 16;
    const int n0 = nt64 * 64;

    const int l15  = lane & 15;
    const int aoff = (lane >> 4) << 3;
    const int boff = (lane >> 4) << 4;

    const _Float16* Q = Qh + (size_t)bh * SEQ * IDIM;
    const _Float16* K = Kh + (size_t)bh * SEQ * IDIM;

    // ---- A fragments (q rows m0..m0+15, K-chunks d=0..31 and 32..63) ----
    const _Float16* qrow = Q + (size_t)(m0 + l15) * IDIM;
    v16h a0, a1;
#pragma unroll
    for (int j = 0; j < 8; ++j) {
        a0[j]     = qrow[boff + j];
        a0[j + 8] = qrow[boff + 8 + j];
        a1[j]     = qrow[32 + boff + j];
        a1[j + 8] = qrow[32 + boff + 8 + j];
    }

    v8f acc[4] = {v8f{}, v8f{}, v8f{}, v8f{}};
#pragma unroll
    for (int nt = 0; nt < 4; ++nt) {
        const int n = n0 + nt * 16 + l15;
        const _Float16* kp = K + (size_t)n * IDIM + boff;
        v16h b0, b1;
#pragma unroll
        for (int j = 0; j < 16; ++j) { b0[j] = kp[j]; b1[j] = kp[32 + j]; }
        acc[nt] = __builtin_amdgcn_wmma_f32_16x16x32_f16(false, a0, false, b0,
                                                         (short)0, acc[nt], false, false);
        acc[nt] = __builtin_amdgcn_wmma_f32_16x16x32_f16(false, a1, false, b1,
                                                         (short)0, acc[nt], false, false);
    }

    // ---- mask + scale + store (lane owns one column n of 8 rows per acc) ----
    const float scale = 0.125f;                  // 1/sqrt(64)
#pragma unroll
    for (int nt = 0; nt < 4; ++nt) {
        const int n   = n0 + nt * 16 + l15;
        const float pad = amask[b * SEQ + n];
        const float padBias = (1.0f - pad) * NEGBIG;
#pragma unroll
        for (int v = 0; v < 8; ++v) {
            const int m = m0 + v + aoff;
            float val = acc[nt][v] * pad - padBias;
            if (n < m) val -= NEGBIG;            // tril(ones, -1) mask
            out[((size_t)bh * SEQ + m) * SEQ + n] = val * scale;
        }
    }
}

// ---------------------------------------------------------------------------
// Launch: inputs are (x f32, attention_mask f32, boxes i32, W f32, b f32)
// Workspace layout (16.5 MB, all L2-resident):
//   cosT 512KB | sinT 512KB | Qh 4MB | Kh 4MB | Xh 6MB | Wh 1.5MB
// ---------------------------------------------------------------------------
extern "C" void kernel_launch(void* const* d_in, const int* in_sizes, int n_in,
                              void* d_out, int out_size, void* d_ws, size_t ws_size,
                              hipStream_t stream) {
    const float* x     = (const float*)d_in[0];
    const float* amask = (const float*)d_in[1];
    const int*   boxes = (const int*)d_in[2];
    const float* W     = (const float*)d_in[3];
    const float* bias  = (const float*)d_in[4];
    float*       out   = (float*)d_out;

    float*    cosT = (float*)d_ws;
    float*    sinT = cosT + (size_t)BATCH * SEQ * 32;
    _Float16* Qh   = (_Float16*)(sinT + (size_t)BATCH * SEQ * 32);
    _Float16* Kh   = Qh + (size_t)BATCH * NLAB * SEQ * IDIM;
    _Float16* Xh   = Kh + (size_t)BATCH * NLAB * SEQ * IDIM;
    _Float16* Wh   = Xh + (size_t)BATCH * SEQ * HID;

    // K0: one-shot f16 copies of x (3.1M elems) and W (0.79M elems)
    {
        const int nx8 = (BATCH * SEQ * HID) / 8;        // 393216
        const int nw8 = (2 * NLAB * IDIM * HID) / 8;    // 98304
        cvt_f16_kernel<<<(nx8 + 255) / 256, 256, 0, stream>>>(x, Xh, nx8);
        cvt_f16_kernel<<<(nw8 + 255) / 256, 256, 0, stream>>>(W, Wh, nw8);
    }

    // K1: 4 blocks x 1024 threads — box ids + sin/cos tables
    box_rope_kernel<<<BATCH, SEQ, 0, stream>>>(boxes, cosT, sinT);

    // K2: 4096 wave-tiles (16 rows x one 64-col chunk), 8 waves/block
    proj_rope_kernel<<<512, 256, 0, stream>>>(Xh, Wh, bias, cosT, sinT, Qh, Kh);

    // K3: 32768 wave-tiles (16x64 logits), 8 waves/block
    logits_kernel<<<4096, 256, 0, stream>>>(Qh, Kh, amask, out);
}
